// GAT_53498112639141
// MI455X (gfx1250) — matrix operand order repaired and checked by reading, not compiled, not run on hardware
//
#include <hip/hip_runtime.h>
#include <hip/hip_bf16.h>

typedef __attribute__((ext_vector_type(16))) __bf16 v16bf;
typedef __attribute__((ext_vector_type(8)))  float  v8f;
typedef __attribute__((ext_vector_type(4)))  int    v4i;

#define NN 20000
#define EE 160000
#define GG 64
#define HH 3
#define MPAD 20096            // 157 * 128
#define NEG_SLOPE 0.2f
#define BN_EPS 1e-5f

#define AS1 __attribute__((address_space(1)))
#define AS3 __attribute__((address_space(3)))

#if defined(__has_builtin)
#if __has_builtin(__builtin_amdgcn_global_load_async_to_lds_b128)
#define HAVE_ASYNC_LDS 1
#endif
#if __has_builtin(__builtin_amdgcn_s_wait_asynccnt)
#define HAVE_WAIT_ASYNC 1
#endif
#endif

// ---------------------------------------------------------------------------
// fp32 -> bf16 convert with zero padding (rows/cols)
// ---------------------------------------------------------------------------
__global__ void k_cvt_pad(const float* __restrict__ src, unsigned short* __restrict__ dst,
                          int rows_src, int cols_src, int rows_dst, int cols_dst)
{
    long long idx = (long long)blockIdx.x * blockDim.x + threadIdx.x;
    long long total = (long long)rows_dst * cols_dst;
    if (idx >= total) return;
    int r = (int)(idx / cols_dst), c = (int)(idx % cols_dst);
    float v = (r < rows_src && c < cols_src) ? src[(long long)r * cols_src + c] : 0.f;
    __hip_bfloat16 b = __float2bfloat16(v);
    dst[idx] = *(unsigned short*)&b;
}

// ---------------------------------------------------------------------------
// WMMA bf16 GEMM: C[MPAD x Ncols] = A[MPAD x K] * B[K x Ncols]
// Block tile 128x128, 8 waves, each wave 32x64 via 2x4 16x16x32 WMMA tiles.
// A staged via async global->LDS (when available); B staged transposed.
// Both LDS tiles use 80-byte row pitch: 16B-aligned rows, conflict-free
// (20*n mod 64 banks cycles through 16 distinct banks for 16 lanes).
// ---------------------------------------------------------------------------
#define A_STRIDE 40   // ushort units -> 80B rows
#define B_STRIDE 40   // ushort units (transposed B: [n][k]) -> 80B rows

__global__ __launch_bounds__(256) void k_gemm_bf16(
    const unsigned short* __restrict__ A, int lda,
    const unsigned short* __restrict__ B, int ldb,
    float* __restrict__ C, int ldc, int K)
{
    __shared__ __align__(16) unsigned short Alds[128][A_STRIDE];
    __shared__ __align__(16) unsigned short Blds[128][B_STRIDE];   // Blds[n][k]

    int tid  = threadIdx.x;
    int lane = tid & 31;
    int wave = tid >> 5;
    int wm = (wave & 3) * 32;       // wave M offset in block tile
    int wn = (wave >> 2) * 64;      // wave N offset in block tile
    long long tileM = (long long)blockIdx.y * 128;
    int tileN = blockIdx.x * 128;

    v8f acc[2][4];
    #pragma unroll
    for (int i = 0; i < 2; i++)
        #pragma unroll
        for (int j = 0; j < 4; j++)
            #pragma unroll
            for (int r = 0; r < 8; r++) acc[i][j][r] = 0.f;

    int arow = tid >> 1, ahalf = (tid & 1) * 16;    // A: 128 rows x 32 cols
    int brow = tid >> 3, bseg  = (tid & 7) * 16;    // B: 32 rows x 128 cols

    const int lm = lane & 15;
    const int lk = (lane >> 4) * 8;      // A frag K group
    const int kb = (lane >> 4) * 8;      // B frag K group

    const unsigned short* Arowp = A + (tileM + arow) * lda + ahalf;

    for (int k0 = 0; k0 < K; k0 += 32) {
        __syncthreads();
        // ---- stage A tile (row-major), async global->LDS if available ----
#ifdef HAVE_ASYNC_LDS
        {
            unsigned short* anc = const_cast<unsigned short*>(Arowp + k0);
            __builtin_amdgcn_global_load_async_to_lds_b128(
                (AS1 v4i*)anc, (AS3 v4i*)&Alds[arow][ahalf], 0, 0);
            __builtin_amdgcn_global_load_async_to_lds_b128(
                (AS1 v4i*)(anc + 8), (AS3 v4i*)&Alds[arow][ahalf + 8], 0, 0);
        }
#else
        {
            const uint4* ga = (const uint4*)(Arowp + k0);
            uint4 a0 = ga[0], a1 = ga[1];
            *(uint4*)&Alds[arow][ahalf]     = a0;
            *(uint4*)&Alds[arow][ahalf + 8] = a1;
        }
#endif
        // ---- stage B tile transposed via VGPRs ----
        union { uint4 q[2]; unsigned short s[16]; } bv;
        const uint4* gb = (const uint4*)(B + (long long)(k0 + brow) * ldb + tileN + bseg);
        bv.q[0] = gb[0]; bv.q[1] = gb[1];
        #pragma unroll
        for (int i = 0; i < 16; i++) Blds[bseg + i][brow] = bv.s[i];
        // prefetch next K tile of B
        if (k0 + 32 < K)
            __builtin_prefetch(B + (long long)(k0 + 32 + brow) * ldb + tileN + bseg, 0, 1);
#ifdef HAVE_ASYNC_LDS
#ifdef HAVE_WAIT_ASYNC
        __builtin_amdgcn_s_wait_asynccnt(0);
#else
        asm volatile("s_wait_asynccnt 0x0" ::: "memory");
#endif
#endif
        __syncthreads();

        // ---- fragment loads (all 16B-aligned contiguous -> ds_load_b128) ----
        union Frag { v16bf v; uint4 q[2]; unsigned u[8]; };
        Frag af[2], bf[4];
        #pragma unroll
        for (int t = 0; t < 2; t++) {
            int m = wm + t * 16 + lm;
            af[t].q[0] = *(const uint4*)&Alds[m][lk];        // K = lk..lk+7
            af[t].q[1] = *(const uint4*)&Alds[m][16 + lk];   // K = 16+lk..16+lk+7
        }
        #pragma unroll
        for (int t = 0; t < 4; t++) {
            int n = wn + t * 16 + lm;
            bf[t].q[0] = *(const uint4*)&Blds[n][kb];        // K = kb..kb+7
            bf[t].q[1] = *(const uint4*)&Blds[n][16 + kb];   // K = 16+kb..16+kb+7
        }
        #pragma unroll
        for (int i = 0; i < 2; i++)
            #pragma unroll
            for (int j = 0; j < 4; j++)
                acc[i][j] = __builtin_amdgcn_wmma_f32_16x16x32_bf16(
                    false, af[i].v, false, bf[j].v, (short)0, acc[i][j], false, false);
    }

    // epilogue: C/D layout -> VGPR r holds M = r + (lane>=16 ? 8 : 0), N = lane&15
    int mo = (lane >> 4) * 8;
    #pragma unroll
    for (int i = 0; i < 2; i++)
        #pragma unroll
        for (int j = 0; j < 4; j++)
            #pragma unroll
            for (int r = 0; r < 8; r++) {
                long long row = tileM + wm + i * 16 + mo + r;
                int col = tileN + wn + j * 16 + lm;
                C[row * ldc + col] = acc[i][j][r];
            }
}

// ---------------------------------------------------------------------------
// Per-node attention scores: es[n,h] = <h[n,h,:], a_src[h,:]>, ed similarly
// ---------------------------------------------------------------------------
__global__ __launch_bounds__(256) void k_scores(const float* __restrict__ h,
    const float* __restrict__ as, const float* __restrict__ ad,
    float* __restrict__ es, float* __restrict__ ed, int C)
{
    __shared__ float r1[256], r2[256];
    int n = blockIdx.x, tid = threadIdx.x, F = HH * C;
    for (int hh = 0; hh < HH; hh++) {
        float p1 = 0.f, p2 = 0.f;
        for (int c = tid; c < C; c += 256) {
            float v = h[(long long)n * F + hh * C + c];
            p1 += v * as[hh * C + c];
            p2 += v * ad[hh * C + c];
        }
        r1[tid] = p1; r2[tid] = p2;
        __syncthreads();
        for (int s = 128; s > 0; s >>= 1) {
            if (tid < s) { r1[tid] += r1[tid + s]; r2[tid] += r2[tid + s]; }
            __syncthreads();
        }
        if (tid == 0) { es[n * HH + hh] = r1[0]; ed[n * HH + hh] = r2[0]; }
        __syncthreads();
    }
}

// order-preserving float <-> uint for atomicMax
__device__ __forceinline__ unsigned fenc(float f) {
    unsigned u = __float_as_uint(f);
    return (u & 0x80000000u) ? ~u : (u | 0x80000000u);
}
__device__ __forceinline__ float fdec(unsigned u) {
    return __uint_as_float((u & 0x80000000u) ? (u & 0x7fffffffu) : ~u);
}

__global__ void k_edge_max(const int* __restrict__ ei, const float* __restrict__ es,
    const float* __restrict__ ed, float* __restrict__ sc, unsigned* __restrict__ menc)
{
    long long idx = (long long)blockIdx.x * blockDim.x + threadIdx.x;
    long long total = (long long)(EE + NN) * HH;
    if (idx >= total) return;
    int e = (int)(idx / HH), hh = (int)(idx % HH);
    int s = (e < EE) ? ei[e] : (e - EE);
    int d = (e < EE) ? ei[EE + e] : (e - EE);
    float v = es[s * HH + hh] + ed[d * HH + hh];
    v = (v > 0.f) ? v : NEG_SLOPE * v;       // leaky_relu
    sc[idx] = v;
    atomicMax(&menc[d * HH + hh], fenc(v));
}

__global__ void k_edge_exp(const int* __restrict__ ei, float* __restrict__ ex,
    const unsigned* __restrict__ menc, float* __restrict__ svec)
{
    long long idx = (long long)blockIdx.x * blockDim.x + threadIdx.x;
    long long total = (long long)(EE + NN) * HH;
    if (idx >= total) return;
    int e = (int)(idx / HH), hh = (int)(idx % HH);
    int d = (e < EE) ? ei[EE + e] : (e - EE);
    float v = __expf(ex[idx] - fdec(menc[d * HH + hh]));
    ex[idx] = v;
    atomicAdd(&svec[d * HH + hh], v);
}

__global__ __launch_bounds__(256) void k_edge_agg(const int* __restrict__ ei,
    const float* __restrict__ ex, const float* __restrict__ svec,
    const float* __restrict__ h, float* __restrict__ out, int C)
{
    int e = blockIdx.x, F = HH * C;
    int s = (e < EE) ? ei[e] : (e - EE);
    int d = (e < EE) ? ei[EE + e] : (e - EE);
    __shared__ float alpha[HH];
    if (threadIdx.x < HH)
        alpha[threadIdx.x] = ex[e * HH + threadIdx.x] / (svec[d * HH + threadIdx.x] + 1e-16f);
    __syncthreads();
    for (int ch = threadIdx.x; ch < F; ch += 256) {
        int hh = ch / C;
        atomicAdd(&out[(long long)d * F + ch], alpha[hh] * h[(long long)s * F + ch]);
    }
}

// ---------------------------------------------------------------------------
// BatchNorm over N rows of relu(hout + bias): stats via partial-sum atomics
// ---------------------------------------------------------------------------
__global__ __launch_bounds__(256) void k_bn_stats(const float* __restrict__ hout,
    const float* __restrict__ bias, float* __restrict__ sums, float* __restrict__ sumsq, int F)
{
    int r0 = blockIdx.x * 128;
    int r1 = r0 + 128; if (r1 > NN) r1 = NN;
    float acc[3] = {0.f, 0.f, 0.f}, acc2[3] = {0.f, 0.f, 0.f};
    for (int r = r0; r < r1; r++) {
        int j = 0;
        for (int ch = threadIdx.x; ch < F; ch += 256, j++) {
            float v = hout[(long long)r * F + ch] + bias[ch];
            v = v > 0.f ? v : 0.f;
            acc[j] += v; acc2[j] += v * v;
        }
    }
    int j = 0;
    for (int ch = threadIdx.x; ch < F; ch += 256, j++) {
        atomicAdd(&sums[ch], acc[j]);
        atomicAdd(&sumsq[ch], acc2[j]);
    }
}

__global__ void k_bn_final(const float* __restrict__ sums, const float* __restrict__ sumsq,
                           float* __restrict__ mean, float* __restrict__ rstd, int F)
{
    int ch = blockIdx.x * 256 + threadIdx.x;
    if (ch >= F) return;
    float m = sums[ch] / (float)NN;
    float var = sumsq[ch] / (float)NN - m * m;
    mean[ch] = m;
    rstd[ch] = rsqrtf(var + BN_EPS);
}

__global__ void k_bn_apply(const float* __restrict__ hout, const float* __restrict__ bias,
    const float* __restrict__ mean, const float* __restrict__ rstd,
    const float* __restrict__ gam, const float* __restrict__ bet,
    unsigned short* __restrict__ dst, int F)
{
    long long idx = (long long)blockIdx.x * blockDim.x + threadIdx.x;
    long long total = (long long)MPAD * F;
    if (idx >= total) return;
    int ch = (int)(idx % F);
    long long row = idx / F;
    float y = 0.f;
    if (row < NN) {
        float v = hout[idx] + bias[ch];
        v = v > 0.f ? v : 0.f;
        y = (v - mean[ch]) * rstd[ch] * gam[ch] + bet[ch];
    }
    __hip_bfloat16 b = __float2bfloat16(y);
    dst[idx] = *(unsigned short*)&b;
}

// ---------------------------------------------------------------------------
// global_add_pool over batch ids
// ---------------------------------------------------------------------------
__global__ void k_pool(const unsigned short* __restrict__ hb, const int* __restrict__ batch,
                       float* __restrict__ pool)
{
    long long idx = (long long)blockIdx.x * blockDim.x + threadIdx.x;
    long long total = (long long)NN * 384;
    if (idx >= total) return;
    int n = (int)(idx / 384), ch = (int)(idx % 384);
    __hip_bfloat16 b; *(unsigned short*)&b = hb[idx];
    atomicAdd(&pool[batch[n] * 384 + ch], __bfloat162float(b));
}

// ---------------------------------------------------------------------------
// MLP head: fc1 + relu + BN + fc2 + relu + fc3 + relu, one block
// ---------------------------------------------------------------------------
__global__ __launch_bounds__(256) void k_head(const float* __restrict__ p,
    const float* __restrict__ fc1w, const float* __restrict__ fc1b,
    const float* __restrict__ g4, const float* __restrict__ be4,
    const float* __restrict__ fc2w, const float* __restrict__ fc2b,
    const float* __restrict__ fc3w, const float* __restrict__ fc3b,
    float* __restrict__ out)
{
    __shared__ float h1[GG][64];
    __shared__ float h2[GG][64];
    __shared__ float mu[64], rs[64];
    int tid = threadIdx.x;
    for (int idx = tid; idx < GG * 64; idx += 256) {
        int g = idx >> 6, j = idx & 63;
        float acc = fc1b[j];
        for (int k = 0; k < 384; k++) acc += p[g * 384 + k] * fc1w[k * 64 + j];
        h1[g][j] = acc > 0.f ? acc : 0.f;
    }
    __syncthreads();
    if (tid < 64) {
        float s = 0.f, s2 = 0.f;
        for (int g = 0; g < GG; g++) { float v = h1[g][tid]; s += v; s2 += v * v; }
        float m = s / (float)GG;
        mu[tid] = m;
        rs[tid] = rsqrtf(s2 / (float)GG - m * m + BN_EPS);
    }
    __syncthreads();
    for (int idx = tid; idx < GG * 64; idx += 256) {
        int g = idx >> 6, j = idx & 63;
        h1[g][j] = (h1[g][j] - mu[j]) * rs[j] * g4[j] + be4[j];
    }
    __syncthreads();
    for (int idx = tid; idx < GG * 64; idx += 256) {
        int g = idx >> 6, j = idx & 63;
        float acc = fc2b[j];
        for (int k = 0; k < 64; k++) acc += h1[g][k] * fc2w[k * 64 + j];
        h2[g][j] = acc > 0.f ? acc : 0.f;
    }
    __syncthreads();
    if (tid < GG) {
        float acc = fc3b[0];
        for (int j = 0; j < 64; j++) acc += h2[tid][j] * fc3w[j];
        out[tid] = acc > 0.f ? acc : 0.f;
    }
}

// ---------------------------------------------------------------------------
extern "C" void kernel_launch(void* const* d_in, const int* in_sizes, int n_in,
                              void* d_out, int out_size, void* d_ws, size_t ws_size,
                              hipStream_t stream)
{
    (void)in_sizes; (void)n_in; (void)out_size; (void)ws_size;
    const float* x     = (const float*)d_in[0];
    const int*   ei    = (const int*)  d_in[1];
    const int*   batch = (const int*)  d_in[2];
    const float* W1 = (const float*)d_in[3];
    const float* a1s = (const float*)d_in[4];
    const float* a1d = (const float*)d_in[5];
    const float* b1  = (const float*)d_in[6];
    const float* W2 = (const float*)d_in[7];
    const float* a2s = (const float*)d_in[8];
    const float* a2d = (const float*)d_in[9];
    const float* b2  = (const float*)d_in[10];
    const float* W3 = (const float*)d_in[11];
    const float* a3s = (const float*)d_in[12];
    const float* a3d = (const float*)d_in[13];
    const float* b3  = (const float*)d_in[14];
    const float* g1 = (const float*)d_in[15];  const float* be1 = (const float*)d_in[16];
    const float* g2 = (const float*)d_in[17];  const float* be2 = (const float*)d_in[18];
    const float* g3 = (const float*)d_in[19];  const float* be3 = (const float*)d_in[20];
    const float* g4 = (const float*)d_in[21];  const float* be4 = (const float*)d_in[22];
    const float* fc1w = (const float*)d_in[23]; const float* fc1b = (const float*)d_in[24];
    const float* fc2w = (const float*)d_in[25]; const float* fc2b = (const float*)d_in[26];
    const float* fc3w = (const float*)d_in[27]; const float* fc3b = (const float*)d_in[28];
    float* out = (float*)d_out;

    // workspace carve-out
    char* w = (char*)d_ws;
    auto take = [&](size_t bytes) -> char* {
        char* ptr = w;
        w += (bytes + 255) & ~(size_t)255;
        return ptr;
    };
    unsigned short* in_bf = (unsigned short*)take((size_t)MPAD * 768 * 2);
    unsigned short* w1bf  = (unsigned short*)take((size_t)384 * 768 * 2);
    unsigned short* w2bf  = (unsigned short*)take((size_t)768 * 384 * 2);
    unsigned short* w3bf  = (unsigned short*)take((size_t)384 * 384 * 2);
    float* hlin  = (float*)take((size_t)MPAD * 768 * 4);
    float* hout  = (float*)take((size_t)NN * 768 * 4);
    float* es    = (float*)take((size_t)NN * HH * 4);
    float* ed    = (float*)take((size_t)NN * HH * 4);
    unsigned* menc = (unsigned*)take((size_t)NN * HH * 4);
    float* svec  = (float*)take((size_t)NN * HH * 4);
    float* exbuf = (float*)take((size_t)(EE + NN) * HH * 4);
    float* sums  = (float*)take(768 * 4);
    float* sumsq = (float*)take(768 * 4);
    float* meanv = (float*)take(768 * 4);
    float* rstdv = (float*)take(768 * 4);
    float* pool  = (float*)take((size_t)GG * 384 * 4);

    auto cdiv = [](long long a, long long b) -> unsigned { return (unsigned)((a + b - 1) / b); };

    // convert inputs/weights to bf16 with padding
    k_cvt_pad<<<cdiv((long long)MPAD * 384, 256), 256, 0, stream>>>(x, in_bf, NN, 373, MPAD, 384);
    k_cvt_pad<<<cdiv(384 * 768, 256), 256, 0, stream>>>(W1, w1bf, 373, 768, 384, 768);
    k_cvt_pad<<<cdiv(768 * 384, 256), 256, 0, stream>>>(W2, w2bf, 768, 384, 768, 384);
    k_cvt_pad<<<cdiv(384 * 384, 256), 256, 0, stream>>>(W3, w3bf, 384, 384, 384, 384);

    auto layer = [&](int K, int Fout, int C, const unsigned short* wbf,
                     const float* as_, const float* ad_, const float* bias,
                     const float* gam, const float* bet)
    {
        dim3 grid(Fout / 128, MPAD / 128);
        k_gemm_bf16<<<grid, 256, 0, stream>>>(in_bf, K, wbf, Fout, hlin, Fout, K);
        k_scores<<<NN, 256, 0, stream>>>(hlin, as_, ad_, es, ed, C);
        (void)hipMemsetAsync(menc, 0, (size_t)NN * HH * 4, stream);   // fenc(-inf) < any real enc; self-loops cover all dsts
        (void)hipMemsetAsync(svec, 0, (size_t)NN * HH * 4, stream);
        (void)hipMemsetAsync(hout, 0, (size_t)NN * Fout * 4, stream);
        long long tot = (long long)(EE + NN) * HH;
        k_edge_max<<<cdiv(tot, 256), 256, 0, stream>>>(ei, es, ed, exbuf, menc);
        k_edge_exp<<<cdiv(tot, 256), 256, 0, stream>>>(ei, exbuf, menc, svec);
        k_edge_agg<<<EE + NN, 256, 0, stream>>>(ei, exbuf, svec, hlin, hout, C);
        (void)hipMemsetAsync(sums, 0, Fout * 4, stream);
        (void)hipMemsetAsync(sumsq, 0, Fout * 4, stream);
        k_bn_stats<<<cdiv(NN, 128), 256, 0, stream>>>(hout, bias, sums, sumsq, Fout);
        k_bn_final<<<cdiv(Fout, 256), 256, 0, stream>>>(sums, sumsq, meanv, rstdv, Fout);
        k_bn_apply<<<cdiv((long long)MPAD * Fout, 256), 256, 0, stream>>>(
            hout, bias, meanv, rstdv, gam, bet, in_bf, Fout);
    };

    layer(384, 768, 256, w1bf, a1s, a1d, b1, g1, be1);   // GAT1: 373(pad 384) -> 3x256
    layer(768, 384, 128, w2bf, a2s, a2d, b2, g2, be2);   // GAT2: 768 -> 3x128
    layer(384, 384, 128, w3bf, a3s, a3d, b3, g3, be3);   // GAT3: 384 -> 3x128

    (void)hipMemsetAsync(pool, 0, (size_t)GG * 384 * 4, stream);
    k_pool<<<cdiv((long long)NN * 384, 256), 256, 0, stream>>>(in_bf, batch, pool);
    k_head<<<1, 256, 0, stream>>>(pool, fc1w, fc1b, g4, be4, fc2w, fc2b, fc3w, fc3b, out);
}
